// DMSAttentionWrapper_8048768713557
// MI455X (gfx1250) — matrix operand order repaired and checked by reading, not compiled, not run on hardware
//
#include <hip/hip_runtime.h>
#include <hip/hip_bf16.h>

// ---------------------------------------------------------------------------
// Problem constants (from the reference): B=2, T=2048, D=2048, H=16, HD=128
// ---------------------------------------------------------------------------
#define BB   2
#define TT   2048
#define DD   2048
#define HH   16
#define HDD  128
#define BT   (BB * TT)          // 4096 rows
#define WIN_ 256
#define TAU_ 0.1f
#define SCALE_ 0.08838834764831845f   // 1/sqrt(128)

typedef __attribute__((ext_vector_type(16))) _Float16 v16h;
typedef __attribute__((ext_vector_type(8)))  _Float16 v8h;
typedef __attribute__((ext_vector_type(8)))  float    v8f;
typedef __attribute__((ext_vector_type(4)))  int      v4i;

// Async global->LDS copy support (gfx1250 GLOBAL_LOAD_ASYNC_TO_LDS_B128)
#if defined(__has_builtin)
#  if __has_builtin(__builtin_amdgcn_global_load_async_to_lds_b128)
#    define USE_ASYNC_COPY 1
#  endif
#endif
#ifndef USE_ASYNC_COPY
#  define USE_ASYNC_COPY 0
#endif

// Address-space-qualified v4i for the async-copy builtin:
//   param0: v4i in AS1 (global source), param1: v4i in AS3 (LDS dest)
typedef __attribute__((address_space(1))) v4i as1_v4i;
typedef __attribute__((address_space(3))) v4i as3_v4i;

// ---------------------------------------------------------------------------
// WMMA helpers (CDNA5 16x16x32 f16 -> f32)
// ---------------------------------------------------------------------------
__device__ __forceinline__ v8f wmma_f16(v16h a, v16h b, v8f c) {
  return __builtin_amdgcn_wmma_f32_16x16x32_f16(
      /*neg_a=*/false, a, /*neg_b=*/false, b,
      /*c_mod=*/(short)0, c, /*reuse_a=*/false, /*reuse_b=*/false);
}

// A-matrix fragment, 16x32 f16 (ISA 7.12.2): lane L holds row M=L&15;
// lanes 0-15 hold K=[k0..k0+7] and [k0+16..k0+23]; lanes 16-31 the +8 halves.
__device__ __forceinline__ v16h load_frag_a(const _Float16* base, int ld,
                                            int m0, int k0, int lane) {
  int row = m0 + (lane & 15);
  int kb  = k0 + ((lane >> 4) << 3);          // 0 or 8
  const _Float16* p = base + (size_t)row * (size_t)ld + kb;
  union { v16h v; v8h h[2]; } u;
  u.h[0] = *reinterpret_cast<const v8h*>(p);
  u.h[1] = *reinterpret_cast<const v8h*>(p + 16);
  return u.v;
}

// B-matrix fragment, 32x16 f16, from a K-contiguous ("Bt", N x K row-major)
// buffer: lane L holds column N=n0+(L&15); lanes 0-15 hold K=[k0..k0+7] and
// [k0+16..k0+23], lanes 16-31 the +8 halves (same K order as load_frag_a).
__device__ __forceinline__ v16h load_frag_b(const _Float16* base, int ld,
                                            int n0, int k0, int lane) {
  int row = n0 + (lane & 15);
  int kb  = k0 + ((lane >> 4) << 3);          // 0 or 8
  const _Float16* p = base + (size_t)row * (size_t)ld + kb;
  union { v16h v; v8h h[2]; } u;
  u.h[0] = *reinterpret_cast<const v8h*>(p);
  u.h[1] = *reinterpret_cast<const v8h*>(p + 16);
  return u.v;
}

// One 16-byte global -> LDS chunk, async when available.
__device__ __forceinline__ void copy16(const _Float16* src, _Float16* dst) {
#if USE_ASYNC_COPY
  __builtin_amdgcn_global_load_async_to_lds_b128(
      (as1_v4i*)src, (as3_v4i*)dst, 0, 0);
#else
  *reinterpret_cast<v8h*>(dst) = *reinterpret_cast<const v8h*>(src);
#endif
}

__device__ __forceinline__ void wait_copies() {
#if USE_ASYNC_COPY
#  if defined(__has_builtin) && __has_builtin(__builtin_amdgcn_s_wait_asynccnt)
  __builtin_amdgcn_s_wait_asynccnt(0);
#  else
  asm volatile("s_wait_asynccnt 0" ::: "memory");
#  endif
#endif
}

// ---------------------------------------------------------------------------
// f32 -> f16 elementwise convert
// ---------------------------------------------------------------------------
__global__ __launch_bounds__(256) void convert_f16_kernel(const float* __restrict__ in,
                                                          _Float16* __restrict__ out,
                                                          int n) {
  int i = blockIdx.x * 256 + threadIdx.x;
  if (i < n) out[i] = (_Float16)in[i];
}

// ---------------------------------------------------------------------------
// Weight transpose + convert: Wt[n][k] = (f16) W[k][n], D x D
// ---------------------------------------------------------------------------
__global__ __launch_bounds__(256) void transpose_w_kernel(const float* __restrict__ W,
                                                          _Float16* __restrict__ Wt) {
  __shared__ float tile[32][33];
  int bx = blockIdx.x * 32;   // n
  int by = blockIdx.y * 32;   // k
  int tx = threadIdx.x & 31;
  int ty = threadIdx.x >> 5;  // 0..7
  #pragma unroll
  for (int r = ty; r < 32; r += 8)
    tile[r][tx] = W[(size_t)(by + r) * DD + (bx + tx)];
  __syncthreads();
  #pragma unroll
  for (int r = ty; r < 32; r += 8)
    Wt[(size_t)(bx + r) * DD + (by + tx)] = (_Float16)tile[tx][r];
}

// ---------------------------------------------------------------------------
// Transpose V: vb[b][t][h*HD+d] (f16) -> vt[(b*H+h)][d][t] (f16)
// ---------------------------------------------------------------------------
__global__ __launch_bounds__(256) void transpose_v_kernel(const _Float16* __restrict__ vb,
                                                          _Float16* __restrict__ vt) {
  __shared__ _Float16 tile[32][33];
  int bh = blockIdx.z;
  int b  = bh >> 4;
  int h  = bh & 15;
  int t0 = blockIdx.x * 32;
  int d0 = blockIdx.y * 32;
  int tx = threadIdx.x & 31;
  int ty = threadIdx.x >> 5;
  #pragma unroll
  for (int r = ty; r < 32; r += 8)
    tile[r][tx] = vb[((size_t)b * TT + (t0 + r)) * DD + h * HDD + (d0 + tx)];
  __syncthreads();
  #pragma unroll
  for (int r = ty; r < 32; r += 8)
    vt[((size_t)bh * HDD + (d0 + r)) * TT + (t0 + tx)] = tile[tx][r];
}

// ---------------------------------------------------------------------------
// Decision head: alpha[row] = sigmoid((hs_row . Wd + bd)/TAU)
// ---------------------------------------------------------------------------
__global__ __launch_bounds__(256) void decision_kernel(const float* __restrict__ hs,
                                                       const float* __restrict__ Wd,
                                                       const float* __restrict__ bd,
                                                       float* __restrict__ alpha) {
  int lane = threadIdx.x & 31;
  int row  = blockIdx.x * 8 + (threadIdx.x >> 5);
  const float* hp = hs + (size_t)row * DD;
  float sum = 0.f;
  for (int i = lane; i < DD; i += 32) sum += hp[i] * Wd[i];
  #pragma unroll
  for (int off = 1; off < 32; off <<= 1) sum += __shfl_xor(sum, off, 32);
  if (lane == 0) {
    float logit = (sum + bd[0]) / TAU_;
    alpha[row] = 1.f / (1.f + __expf(-logit));
  }
}

// ---------------------------------------------------------------------------
// GEMM: C[M x N] = A[M x K] (f16 row-major) x Bt[N x K] (f16 K-contiguous).
// Block 256 threads = 8 waves (2x4); wave tile 64x64 (4x4 WMMA); block tile
// 128x256.  Double-buffered LDS staging fed by async global->LDS copies.
// ---------------------------------------------------------------------------
#define GEMM_BM 128
#define GEMM_BN 256
#define GEMM_BK 32
#define LDS_LD  40   // halves per padded LDS row (80B, 16B-aligned, bank skew)

__device__ __forceinline__ void stage_tile(const _Float16* __restrict__ gA,
                                           const _Float16* __restrict__ gB,
                                           _Float16* lA, _Float16* lB,
                                           int K, int k0, int tid) {
  // A tile: BM rows x 32 halves = 512 16B-chunks -> 2 per thread
  #pragma unroll
  for (int c = 0; c < 2; ++c) {
    int chunk = tid * 2 + c;
    int row = chunk >> 2, seg = chunk & 3;
    copy16(gA + (size_t)row * K + k0 + seg * 8, lA + row * LDS_LD + seg * 8);
  }
  // B tile: BN rows x 32 halves = 1024 16B-chunks -> 4 per thread
  #pragma unroll
  for (int c = 0; c < 4; ++c) {
    int chunk = tid * 4 + c;
    int row = chunk >> 2, seg = chunk & 3;
    copy16(gB + (size_t)row * K + k0 + seg * 8, lB + row * LDS_LD + seg * 8);
  }
}

template <typename TOut>
__global__ __launch_bounds__(256) void gemm_wmma_kernel(const _Float16* __restrict__ A,
                                                        const _Float16* __restrict__ Bt,
                                                        TOut* __restrict__ C,
                                                        int M, int N, int K) {
  __shared__ __align__(16) _Float16 Abuf[2][GEMM_BM * LDS_LD];  // 2 x 10KB
  __shared__ __align__(16) _Float16 Bbuf[2][GEMM_BN * LDS_LD];  // 2 x 20KB

  int tid  = threadIdx.x;
  int lane = tid & 31;
  int wave = tid >> 5;                // 0..7
  int wm = wave >> 2;                 // 2 waves along M
  int wn = wave & 3;                  // 4 waves along N

  const _Float16* gA = A  + (size_t)(blockIdx.y * GEMM_BM) * K;
  const _Float16* gB = Bt + (size_t)(blockIdx.x * GEMM_BN) * K;

  v8f acc[4][4] = {};

  stage_tile(gA, gB, Abuf[0], Bbuf[0], K, 0, tid);

  int nk = K / GEMM_BK;
  for (int it = 0; it < nk; ++it) {
    int cur = it & 1;
    wait_copies();          // my async copies into buf[cur] have landed
    __syncthreads();        // everyone's copies landed; prior reads of buf[cur^1] done
    if (it + 1 < nk)
      stage_tile(gA, gB, Abuf[cur ^ 1], Bbuf[cur ^ 1], K, (it + 1) * GEMM_BK, tid);

    v16h af[4], bf[4];
    #pragma unroll
    for (int m = 0; m < 4; ++m)
      af[m] = load_frag_a(Abuf[cur], LDS_LD, wm * 64 + 16 * m, 0, lane);
    #pragma unroll
    for (int n = 0; n < 4; ++n)
      bf[n] = load_frag_b(Bbuf[cur], LDS_LD, wn * 64 + 16 * n, 0, lane);
    #pragma unroll
    for (int m = 0; m < 4; ++m)
      #pragma unroll
      for (int n = 0; n < 4; ++n)
        acc[m][n] = wmma_f16(af[m], bf[n], acc[m][n]);
  }

  int hi = lane >> 4, lo = lane & 15;
  int m0 = blockIdx.y * GEMM_BM + wm * 64;
  int n0 = blockIdx.x * GEMM_BN + wn * 64;
  #pragma unroll
  for (int m = 0; m < 4; ++m)
    #pragma unroll
    for (int n = 0; n < 4; ++n)
      #pragma unroll
      for (int r = 0; r < 8; ++r) {
        int row = m0 + 16 * m + r + hi * 8;   // C layout: VGPR r -> M = r + 8*hi
        int col = n0 + 16 * n + lo;           //           lane  -> N
        C[(size_t)row * N + col] = (TOut)acc[m][n][r];
      }
}

// ---------------------------------------------------------------------------
// Flash attention with DMS mask.
// grid = (T/128, B*H); block = 256 (8 waves); wave owns 16 query rows.
// ---------------------------------------------------------------------------
__global__ __launch_bounds__(256) void attention_kernel(const _Float16* __restrict__ qb,
                                                        const _Float16* __restrict__ kb,
                                                        const _Float16* __restrict__ vt,
                                                        const float* __restrict__ alpha,
                                                        _Float16* __restrict__ attn) {
  __shared__ __align__(16) _Float16 lds_p[8][16][32];  // per-wave P tile

  int lane = threadIdx.x & 31;
  int wave = threadIdx.x >> 5;
  int hi = lane >> 4, lo = lane & 15;

  int bh = blockIdx.y;
  int b  = bh >> 4;
  int h  = bh & 15;
  int qbase = blockIdx.x * 128 + wave * 16;

  const _Float16* Qp  = qb + (size_t)b * TT * DD + h * HDD;   // row stride DD
  const _Float16* Kp  = kb + (size_t)b * TT * DD + h * HDD;   // row stride DD
  const _Float16* Vtp = vt + (size_t)bh * HDD * TT;           // row (=d) stride TT
  const float* alp = alpha + b * TT;

  v16h qf[4];
  #pragma unroll
  for (int i = 0; i < 4; ++i) qf[i] = load_frag_a(Qp, DD, qbase, 32 * i, lane);

  v8f o[8] = {};
  float mrow[8], lrow[8];
  #pragma unroll
  for (int r = 0; r < 8; ++r) { mrow[r] = -1e30f; lrow[r] = 0.f; }

  for (int kvb = 0; kvb < qbase + 16; kvb += 32) {
    v8f s0 = {}, s1 = {};
    #pragma unroll
    for (int i = 0; i < 4; ++i) {
      v16h bk0 = load_frag_b(Kp, DD, kvb,      32 * i, lane);
      v16h bk1 = load_frag_b(Kp, DD, kvb + 16, 32 * i, lane);
      s0 = wmma_f16(qf[i], bk0, s0);
      s1 = wmma_f16(qf[i], bk1, s1);
    }

    int t0 = kvb + lo, t1 = kvb + 16 + lo;
    float a0 = alp[t0], a1 = alp[t1];
    #pragma unroll
    for (int r = 0; r < 8; ++r) {
      int q = qbase + r + hi * 8;
      float x0 = s0[r] * SCALE_;
      float x1 = s1[r] * SCALE_;
      if (t0 > q || (q > t0 + WIN_ && a0 > 0.5f)) x0 = -1e30f;
      if (t1 > q || (q > t1 + WIN_ && a1 > 0.5f)) x1 = -1e30f;
      s0[r] = x0; s1[r] = x1;
    }

    #pragma unroll
    for (int r = 0; r < 8; ++r) {
      float tmax = fmaxf(s0[r], s1[r]);
      #pragma unroll
      for (int off = 1; off < 16; off <<= 1)
        tmax = fmaxf(tmax, __shfl_xor(tmax, off, 32));
      float mnew  = fmaxf(mrow[r], tmax);
      float scale = __expf(mrow[r] - mnew);
      float p0 = (s0[r] <= -1e29f) ? 0.f : __expf(s0[r] - mnew);
      float p1 = (s1[r] <= -1e29f) ? 0.f : __expf(s1[r] - mnew);
      float psum = p0 + p1;
      #pragma unroll
      for (int off = 1; off < 16; off <<= 1)
        psum += __shfl_xor(psum, off, 32);
      lrow[r] = lrow[r] * scale + psum;
      mrow[r] = mnew;
      #pragma unroll
      for (int j = 0; j < 8; ++j) o[j][r] *= scale;
      s0[r] = p0; s1[r] = p1;
    }

    #pragma unroll
    for (int r = 0; r < 8; ++r) {
      lds_p[wave][r + hi * 8][lo]      = (_Float16)s0[r];
      lds_p[wave][r + hi * 8][16 + lo] = (_Float16)s1[r];
    }
    asm volatile("s_wait_dscnt 0" ::: "memory");
    union { v16h v; v8h hh[2]; } up;
    const _Float16* lp = &lds_p[wave][lo][hi * 8];
    up.hh[0] = *reinterpret_cast<const v8h*>(lp);
    up.hh[1] = *reinterpret_cast<const v8h*>(lp + 16);
    v16h pf = up.v;

    #pragma unroll
    for (int j = 0; j < 8; ++j) {
      v16h bv = load_frag_b(Vtp, TT, 16 * j, kvb, lane);
      o[j] = wmma_f16(pf, bv, o[j]);
    }
  }

  #pragma unroll
  for (int j = 0; j < 8; ++j)
    #pragma unroll
    for (int r = 0; r < 8; ++r) {
      int q = qbase + r + hi * 8;
      int d = 16 * j + lo;
      float val = o[j][r] / lrow[r];
      attn[((size_t)b * TT + q) * DD + h * HDD + d] = (_Float16)val;
    }
}

// ---------------------------------------------------------------------------
// Host-side orchestration
// ---------------------------------------------------------------------------
extern "C" void kernel_launch(void* const* d_in, const int* in_sizes, int n_in,
                              void* d_out, int out_size, void* d_ws, size_t ws_size,
                              hipStream_t stream) {
  (void)in_sizes; (void)n_in; (void)out_size; (void)ws_size;

  const float* hs = (const float*)d_in[0];
  const float* Wq = (const float*)d_in[1];
  const float* Wk = (const float*)d_in[2];
  const float* Wv = (const float*)d_in[3];
  const float* Wo = (const float*)d_in[4];
  const float* Wd = (const float*)d_in[5];
  const float* bd = (const float*)d_in[6];
  float* out = (float*)d_out;

  char* ws = (char*)d_ws;
  size_t off = 0;
  auto alloc = [&](size_t bytes) {
    void* p = ws + off;
    off = (off + bytes + 255) & ~(size_t)255;
    return p;
  };
  const size_t actE = (size_t)BT * DD;
  const size_t wE   = (size_t)DD * DD;
  _Float16* hsb  = (_Float16*)alloc(actE * 2);
  _Float16* wtq  = (_Float16*)alloc(wE * 2);
  _Float16* wtk  = (_Float16*)alloc(wE * 2);
  _Float16* wtv  = (_Float16*)alloc(wE * 2);
  _Float16* wto  = (_Float16*)alloc(wE * 2);
  _Float16* qb   = (_Float16*)alloc(actE * 2);
  _Float16* kbuf = (_Float16*)alloc(actE * 2);
  _Float16* vb   = (_Float16*)alloc(actE * 2);
  _Float16* vt   = (_Float16*)alloc(actE * 2);
  _Float16* attn = (_Float16*)alloc(actE * 2);
  float*    alp  = (float*)alloc((size_t)BT * 4);

  // 1) hidden_states f32 -> f16
  {
    int n = (int)actE;
    convert_f16_kernel<<<(n + 255) / 256, 256, 0, stream>>>(hs, hsb, n);
  }
  // 2) weights: transpose + convert to f16 (N x K, K-contiguous)
  {
    dim3 g(DD / 32, DD / 32);
    transpose_w_kernel<<<g, 256, 0, stream>>>(Wq, wtq);
    transpose_w_kernel<<<g, 256, 0, stream>>>(Wk, wtk);
    transpose_w_kernel<<<g, 256, 0, stream>>>(Wv, wtv);
    transpose_w_kernel<<<g, 256, 0, stream>>>(Wo, wto);
  }
  // 3) decision head -> alpha
  decision_kernel<<<BT / 8, 256, 0, stream>>>(hs, Wd, bd, alp);
  // 4) Q/K/V projections (WMMA GEMM, f16 out)
  {
    dim3 g(DD / GEMM_BN, BT / GEMM_BM);
    gemm_wmma_kernel<_Float16><<<g, 256, 0, stream>>>(hsb, wtq, qb,   BT, DD, DD);
    gemm_wmma_kernel<_Float16><<<g, 256, 0, stream>>>(hsb, wtk, kbuf, BT, DD, DD);
    gemm_wmma_kernel<_Float16><<<g, 256, 0, stream>>>(hsb, wtv, vb,   BT, DD, DD);
  }
  // 5) V transpose per (b,h): [B][T][D] -> [B*H][HD][T]
  {
    dim3 g(TT / 32, HDD / 32, BB * HH);
    transpose_v_kernel<<<g, 256, 0, stream>>>(vb, vt);
  }
  // 6) flash attention with DMS mask
  {
    dim3 g(TT / 128, BB * HH);
    attention_kernel<<<g, 256, 0, stream>>>(qb, kbuf, vt, alp, attn);
  }
  // 7) output projection (WMMA GEMM, f32 out -> d_out)
  {
    dim3 g(DD / GEMM_BN, BT / GEMM_BM);
    gemm_wmma_kernel<float><<<g, 256, 0, stream>>>(attn, wto, out, BT, DD, DD);
  }
}
